// Decoder_80891414053560
// MI455X (gfx1250) — matrix operand order repaired and checked by reading, not compiled, not run on hardware
//
#include <hip/hip_runtime.h>
#include <hip/hip_bf16.h>

// ---------------------------------------------------------------------------
// Decoder forward for MI455X (gfx1250).
//  - All GEMM operands pre-converted to bf16 once (single elementwise pass).
//  - GEMMs: v_wmma_f32_16x16x32_bf16, 128x128 block tile, K-step 32,
//    double-buffered LDS fed by global_load_async_to_lds_b128 (ASYNCcnt).
// ---------------------------------------------------------------------------

#define Bd    64
#define Td    128
#define Sd    256
#define NINd  256
#define NOUTd 256
#define Ed    512
#define Hd    1024
#define Ld    3

typedef __attribute__((ext_vector_type(16))) __bf16 v16bf;
typedef __attribute__((ext_vector_type(8)))  float  v8f;

union Frag16 {            // 16 bf16 = 32 bytes = two uint4
    uint4 q[2];
    v16bf v;
};

__device__ __forceinline__ unsigned short f2bf(float f) {
    unsigned int u = __builtin_bit_cast(unsigned int, f);
    u += 0x7FFFu + ((u >> 16) & 1u);        // round-to-nearest-even
    return (unsigned short)(u >> 16);
}
__device__ __forceinline__ unsigned int pk2(float a, float b) {
    return (unsigned int)f2bf(a) | ((unsigned int)f2bf(b) << 16);
}
__device__ __forceinline__ float bf2f(unsigned short u) {
    return __builtin_bit_cast(float, ((unsigned int)u) << 16);
}
__device__ __forceinline__ float sigm(float x) { return 1.0f / (1.0f + expf(-x)); }

// ---- CDNA5 async copy: global -> LDS, 16B per lane, tracked by ASYNCcnt ----
__device__ __forceinline__ void async_b128(void* lds, const void* g) {
    unsigned la = (unsigned)(unsigned long long)lds;          // low 32 bits = LDS offset
    unsigned long long ga = (unsigned long long)g;
    asm volatile("global_load_async_to_lds_b128 %0, %1, off"
                 :: "v"(la), "v"(ga) : "memory");
}
__device__ __forceinline__ void wait_async0() {
    asm volatile("s_wait_asynccnt 0x0" ::: "memory");
}

// ---------------------------------------------------------------------------
// Tiled WMMA GEMM:  C[M,N] = A[M,K] * W[N,K]^T  (+ epilogue). A,W in bf16.
// 256 threads = 8 waves (2M x 4N); each wave: 4x2 wmma 16x16 tiles (64x32).
// EPI: 0 = bias + fp32 store
//      1 = bias + relu + bf16 store
//      2 = attention scores: tanh(acc + hpart[b,col]) * v[col] -> atomicAdd
//      3 = bias + bf16 store
// ---------------------------------------------------------------------------
template <int EPI>
__global__ __launch_bounds__(256) void gemm_wmma(
    const unsigned short* __restrict__ A, int ldA,
    const unsigned short* __restrict__ Bw, int ldB, int bColOff,
    const float* __restrict__ bias,
    void* __restrict__ Cptr, int ldC, int cColOff,
    int M, int N, int K,
    const float* __restrict__ hpart, const float* __restrict__ vvec,
    float* __restrict__ scores)
{
    const int tid   = threadIdx.x;
    const int lane  = tid & 31;
    const int wave  = tid >> 5;
    const int waveM = wave >> 2;      // 0..1
    const int waveN = wave & 3;       // 0..3
    const int bm = blockIdx.y * 128;
    const int bn = blockIdx.x * 128;

    // double-buffered tiles, padded rows (48 halves = 96B) for 16B-aligned b128
    __shared__ __align__(16) unsigned short As[2][128 * 48];
    __shared__ __align__(16) unsigned short Bs[2][128 * 48];

    v8f acc[4][2];
#pragma unroll
    for (int tm = 0; tm < 4; ++tm)
#pragma unroll
        for (int tn = 0; tn < 2; ++tn)
            acc[tm][tn] = (v8f){0.f, 0.f, 0.f, 0.f, 0.f, 0.f, 0.f, 0.f};

    const int ldr   = tid >> 1;            // 0..127 : tile row staged by this thread
    const int ldc16 = (tid & 1) * 16;      // 0 or 16 : col base (16 bf16 = 32B)
    const int arow  = (bm + ldr < M) ? (bm + ldr) : (M - 1);   // clamp (M=64 case)
    const unsigned short* gA = A  + (long)arow * ldA + ldc16;
    const unsigned short* gB = Bw + (long)(bn + ldr) * ldB + bColOff + ldc16;

    const int KT = K >> 5;

    // issue tile 0
    {
        unsigned short* as = &As[0][ldr * 48 + ldc16];
        unsigned short* bs = &Bs[0][ldr * 48 + ldc16];
        async_b128(as,     gA);
        async_b128(as + 8, gA + 8);
        async_b128(bs,     gB);
        async_b128(bs + 8, gB + 8);
    }

    for (int kt = 0; kt < KT; ++kt) {
        wait_async0();          // this wave's tile-kt copies are in LDS
        __syncthreads();        // everyone's are; previous compute finished

        if (kt + 1 < KT) {      // prefetch tile kt+1 into the other buffer
            const int nb = (kt + 1) & 1;
            const long ko = (long)(kt + 1) << 5;
            unsigned short* as = &As[nb][ldr * 48 + ldc16];
            unsigned short* bs = &Bs[nb][ldr * 48 + ldc16];
            async_b128(as,     gA + ko);
            async_b128(as + 8, gA + ko + 8);
            async_b128(bs,     gB + ko);
            async_b128(bs + 8, gB + ko + 8);
        }

        const int cb = kt & 1;
        // A frag (ISA 16-bit A 16x32): lane<16 -> K {0..7,16..23}; lane>=16 -> {8..15,24..31}
        Frag16 fa[4];
        const int aseg = (lane >> 4) << 3;          // 0 or 8
#pragma unroll
        for (int tm = 0; tm < 4; ++tm) {
            const int r = (waveM * 64 + tm * 16 + (lane & 15)) * 48;
            fa[tm].q[0] = *(const uint4*)&As[cb][r + aseg];
            fa[tm].q[1] = *(const uint4*)&As[cb][r + aseg + 16];
        }
        // B frag (32x16): lanes 0-15 hold K {0..7,16..23} of col n; lanes 16-31 hold K {8..15,24..31}
        Frag16 fb[2];
        const int bseg = (lane >> 4) << 3;          // 0 or 8
#pragma unroll
        for (int tn = 0; tn < 2; ++tn) {
            const int r = (waveN * 32 + tn * 16 + (lane & 15)) * 48 + bseg;
            fb[tn].q[0] = *(const uint4*)&Bs[cb][r];
            fb[tn].q[1] = *(const uint4*)&Bs[cb][r + 16];
        }
#pragma unroll
        for (int tm = 0; tm < 4; ++tm)
#pragma unroll
            for (int tn = 0; tn < 2; ++tn)
                acc[tm][tn] = __builtin_amdgcn_wmma_f32_16x16x32_bf16(
                    false, fa[tm].v, false, fb[tn].v,
                    (short)0, acc[tm][tn], false, false);

        __syncthreads();        // fragments consumed before buffer reuse
    }

    // ---- epilogue ----
    if constexpr (EPI == 2) {
#pragma unroll
        for (int tm = 0; tm < 4; ++tm) {
            const int rowbase = bm + waveM * 64 + tm * 16 + ((lane >> 4) << 3);
#pragma unroll
            for (int i = 0; i < 8; ++i) {
                const int row = rowbase + i;
                const int b   = row >> 8;               // S = 256
                float p = 0.f;
#pragma unroll
                for (int tn = 0; tn < 2; ++tn) {
                    const int col = bn + waveN * 32 + tn * 16 + (lane & 15);
                    float t = tanhf(acc[tm][tn][i] + hpart[(long)b * Hd + col]);
                    p += t * vvec[col];
                }
                for (int off = 1; off < 16; off <<= 1)    // reduce 16 N-lanes/half
                    p += __shfl_xor(p, off, 16);
                if ((lane & 15) == 0)
                    atomicAdd(&scores[row], p);
            }
        }
    } else {
#pragma unroll
        for (int tm = 0; tm < 4; ++tm) {
            const int rowbase = bm + waveM * 64 + tm * 16 + ((lane >> 4) << 3);
#pragma unroll
            for (int tn = 0; tn < 2; ++tn) {
                const int col = bn + waveN * 32 + tn * 16 + (lane & 15);
                const float bv = bias ? bias[col] : 0.f;
#pragma unroll
                for (int i = 0; i < 8; ++i) {
                    const int row = rowbase + i;
                    if (row >= M) continue;
                    float v = acc[tm][tn][i] + bv;
                    if constexpr (EPI == 1) {
                        v = v > 0.f ? v : 0.f;
                        ((unsigned short*)Cptr)[(long)row * ldC + cColOff + col] = f2bf(v);
                    } else if constexpr (EPI == 3) {
                        ((unsigned short*)Cptr)[(long)row * ldC + cColOff + col] = f2bf(v);
                    } else {
                        ((float*)Cptr)[(long)row * ldC + cColOff + col] = v;
                    }
                }
            }
        }
    }
}

// ---------------------------------------------------------------------------
__global__ void zero_f32(float* p, int n) {
    int i = blockIdx.x * blockDim.x + threadIdx.x;
    if (i < n) p[i] = 0.f;
}

// fp32 -> bf16, 4 elements/thread
__global__ void cvt_bf16(const float* __restrict__ s, unsigned short* __restrict__ d, long n4) {
    long i = (long)blockIdx.x * blockDim.x + threadIdx.x;
    if (i < n4) {
        float4 f = ((const float4*)s)[i];
        uint2 u;
        u.x = pk2(f.x, f.y);
        u.y = pk2(f.z, f.w);
        ((uint2*)d)[i] = u;
    }
}

// softmax over S=256 scores per batch, then context[b,h] broadcast (bf16)
// into x0[:, 512:1536] for every timestep of batch b.
__global__ __launch_bounds__(256) void softmax_context(
    const float* __restrict__ scores, const float* __restrict__ enc,
    unsigned short* __restrict__ x0)
{
    const int b = blockIdx.x, tid = threadIdx.x;
    __shared__ float sm[256];
    __shared__ float attn[256];

    const float v = scores[b * Sd + tid];
    sm[tid] = v;
    __syncthreads();
    for (int off = 128; off > 0; off >>= 1) {
        if (tid < off) sm[tid] = fmaxf(sm[tid], sm[tid + off]);
        __syncthreads();
    }
    const float mx = sm[0];
    __syncthreads();
    const float e = expf(v - mx);
    attn[tid] = e;
    sm[tid]   = e;
    __syncthreads();
    for (int off = 128; off > 0; off >>= 1) {
        if (tid < off) sm[tid] += sm[tid + off];
        __syncthreads();
    }
    const float inv = 1.0f / sm[0];

#pragma unroll
    for (int hh = 0; hh < 4; ++hh) {
        const int h = tid + hh * 256;
        const float* ep = enc + (long)b * Sd * Hd + h;
        float c = 0.f;
        for (int s = 0; s < Sd; ++s) c += attn[s] * ep[(long)s * Hd];
        const unsigned short cb = f2bf(c * inv);
        unsigned short* xp = x0 + (long)b * Td * (Ed + Hd) + Ed + h;
        for (int t = 0; t < Td; ++t) xp[(long)t * (Ed + Hd)] = cb;
    }
}

// GRU gate combine: gx (bf16, [row,3H]) + gh (f32, [b,3H]) -> out bf16 [row,H]
__global__ __launch_bounds__(256) void gru_combine(
    const unsigned short* __restrict__ gx, const float* __restrict__ gh,
    const float* __restrict__ h0, unsigned short* __restrict__ out)
{
    const long idx = (long)blockIdx.x * 256 + threadIdx.x;   // 8192*1024 total
    const int row = (int)(idx >> 10);
    const int h   = (int)(idx & 1023);
    const int b   = row >> 7;                                 // T = 128
    const unsigned short* gxr = gx + (long)row * 3072;
    const float* ghr = gh + (long)b * 3072;
    const float r = sigm(bf2f(gxr[h]) + ghr[h]);
    const float z = sigm(bf2f(gxr[1024 + h]) + ghr[1024 + h]);
    const float n = tanhf(bf2f(gxr[2048 + h]) + r * ghr[2048 + h]);
    const float hv = h0[(long)b * Hd + h];
    out[idx] = f2bf((1.f - z) * n + z * hv);
}

// ---------------------------------------------------------------------------
extern "C" void kernel_launch(void* const* d_in, const int* in_sizes, int n_in,
                              void* d_out, int out_size, void* d_ws, size_t ws_size,
                              hipStream_t stream)
{
    (void)in_sizes; (void)n_in; (void)out_size; (void)ws_size;

    const float* dec   = (const float*)d_in[0];   // [B,T,NIN]
    const float* enc   = (const float*)d_in[1];   // [B,S,H]
    const float* eh    = (const float*)d_in[2];   // [L,1,B,H]
    const float* Wemb  = (const float*)d_in[3];   // [E,NIN]
    const float* bemb  = (const float*)d_in[4];
    const float* attnW = (const float*)d_in[5];   // [H,2H]
    const float* attnb = (const float*)d_in[6];
    const float* attnv = (const float*)d_in[7];   // [1,H]
    const float* Wih0  = (const float*)d_in[8];   // [3H,E+H]
    const float* Wihr  = (const float*)d_in[9];   // [L-1,3H,H]
    const float* Whh   = (const float*)d_in[10];  // [L,3H,H]
    const float* bih   = (const float*)d_in[11];  // [L,3H]
    const float* bhh   = (const float*)d_in[12];  // [L,3H]
    const float* fcW   = (const float*)d_in[13];  // [NOUT,H]
    const float* fcb   = (const float*)d_in[14];

    const int BT = Bd * Td;                        // 8192
    char* ws = (char*)d_ws;
    typedef unsigned short u16;
    // ---- workspace layout (bytes) ----
    u16*  x0     = (u16*)(ws +         0);   // [8192,1536] bf16   25,165,824
    u16*  xa     = (u16*)(ws +  25165824);   // [8192,1024] bf16   16,777,216
    u16*  xb     = (u16*)(ws +  41943040);   // [8192,1024] bf16   16,777,216
    u16*  gx     = (u16*)(ws +  58720256);   // [8192,3072] bf16   50,331,648
    float* gh    = (float*)(ws + 109051904); // [64,3072]   f32       786,432
    float* hpart = (float*)(ws + 109838336); // [64,1024]   f32       262,144
    float* scores= (float*)(ws + 110100480); // [64*256]    f32        65,536
    u16*  decb   = (u16*)(ws + 110166016);   // [8192,256]          4,194,304
    u16*  encb   = (u16*)(ws + 114360320);   // [16384,1024]       33,554,432
    u16*  ehb    = (u16*)(ws + 147914752);   // [3,64,1024]           393,216
    u16*  Wembb  = (u16*)(ws + 148307968);   // [512,256]             262,144
    u16*  attnWb = (u16*)(ws + 148570112);   // [1024,2048]         4,194,304
    u16*  Wih0b  = (u16*)(ws + 152764416);   // [3072,1536]         9,437,184
    u16*  Wihrb  = (u16*)(ws + 162201600);   // [2,3072,1024]      12,582,912
    u16*  Whhb   = (u16*)(ws + 174784512);   // [3,3072,1024]      18,874,368
    u16*  fcWb   = (u16*)(ws + 193658880);   // [256,1024]            524,288

    dim3 blk(256);
    auto cvt = [&](const float* s, u16* d, long n) {
        long n4 = n >> 2;
        cvt_bf16<<<dim3((unsigned)((n4 + 255) / 256)), blk, 0, stream>>>(s, d, n4);
    };
    // ---- one-time bf16 conversion of all GEMM operands ----
    cvt(dec,   decb,   (long)BT * NINd);
    cvt(enc,   encb,   (long)Bd * Sd * Hd);
    cvt(eh,    ehb,    (long)Ld * Bd * Hd);
    cvt(Wemb,  Wembb,  (long)Ed * NINd);
    cvt(attnW, attnWb, (long)Hd * 2 * Hd);
    cvt(Wih0,  Wih0b,  (long)3 * Hd * (Ed + Hd));
    cvt(Wihr,  Wihrb,  (long)2 * 3 * Hd * Hd);
    cvt(Whh,   Whhb,   (long)3 * 3 * Hd * Hd);
    cvt(fcW,   fcWb,   (long)NOUTd * Hd);

    // scores <- 0
    zero_f32<<<dim3((Bd * Sd + 255) / 256), blk, 0, stream>>>(scores, Bd * Sd);

    // G1: emb = relu(dec @ Wemb^T + bemb) -> x0[:, 0:512] bf16
    gemm_wmma<1><<<dim3(Ed / 128, BT / 128), blk, 0, stream>>>(
        decb, NINd, Wembb, NINd, 0, bemb, x0, Ed + Hd, 0,
        BT, Ed, NINd, nullptr, nullptr, nullptr);

    // G2: hpart = h_last @ Wa^T + attnb   (Wa = attn_W[:, :H])
    gemm_wmma<0><<<dim3(Hd / 128, 1), blk, 0, stream>>>(
        ehb + (long)(Ld - 1) * Bd * Hd, Hd, attnWb, 2 * Hd, 0, attnb,
        hpart, Hd, 0, Bd, Hd, Hd, nullptr, nullptr, nullptr);

    // G3: scores[b,s] += v . tanh(enc @ Wb^T + hpart[b])   (Wb = attn_W[:, H:])
    gemm_wmma<2><<<dim3(Hd / 128, (Bd * Sd) / 128), blk, 0, stream>>>(
        encb, Hd, attnWb, 2 * Hd, Hd, nullptr, nullptr, 0, 0,
        Bd * Sd, Hd, Hd, hpart, attnv, scores);

    // softmax + context -> x0[:, 512:1536] bf16 (broadcast over T)
    softmax_context<<<dim3(Bd), blk, 0, stream>>>(scores, enc, x0);

    const long whl = (long)3 * Hd * Hd;            // per-layer W stride
    const unsigned int cgrid = (unsigned int)(((long)BT * Hd) / 256);

    const u16*  xin[3]  = { x0, xa, xb };
    u16*        xout[3] = { xa, xb, xa };
    const u16*  wih[3]  = { Wih0b, Wihrb, Wihrb + whl };
    const int   kin[3]  = { Ed + Hd, Hd, Hd };

    for (int l = 0; l < Ld; ++l) {
        // gh = h_l @ W_hh[l]^T + b_hh[l]
        gemm_wmma<0><<<dim3(3 * Hd / 128, 1), blk, 0, stream>>>(
            ehb + (long)l * Bd * Hd, Hd, Whhb + l * whl, Hd, 0, bhh + l * 3 * Hd,
            gh, 3 * Hd, 0, Bd, 3 * Hd, Hd, nullptr, nullptr, nullptr);
        // gx = x @ W_ih[l]^T + b_ih[l]   (bf16 out)
        gemm_wmma<3><<<dim3(3 * Hd / 128, BT / 128), blk, 0, stream>>>(
            xin[l], kin[l], wih[l], kin[l], 0, bih + l * 3 * Hd,
            gx, 3 * Hd, 0, BT, 3 * Hd, kin[l], nullptr, nullptr, nullptr);
        // gates
        gru_combine<<<dim3(cgrid), blk, 0, stream>>>(
            gx, gh, eh + (long)l * Bd * Hd, xout[l]);
    }

    // fc_out: d_out = x @ fcW^T + fcb   (fp32 output)
    gemm_wmma<0><<<dim3(NOUTd / 128, BT / 128), blk, 0, stream>>>(
        xa, Hd, fcWb, Hd, 0, fcb,
        d_out, NOUTd, 0, BT, NOUTd, Hd, nullptr, nullptr, nullptr);
}
